// GEARS_Model_52613349376591
// MI455X (gfx1250) — compile-verified, matrix operand order checked by reading, not run on hardware
//
#include <hip/hip_runtime.h>

typedef _Float16 h16;
typedef __attribute__((ext_vector_type(8)))  h16   v8h;
typedef __attribute__((ext_vector_type(16))) h16   v16h;
typedef __attribute__((ext_vector_type(8)))  float v8f;

#define GG 5000
#define PP 2000
#define BB 64
#define HH 64
#define WPAD 5024   // K-pad for cross-gene GEMM (multiple of 32)
#define NPAD 5008   // N-pad for final decoder GEMM (multiple of 16)

// ---------------------------------------------------------------------------
// WMMA helpers (CDNA5 gfx1250, wave32)
// ---------------------------------------------------------------------------
__device__ __forceinline__ v8f wmma16(v16h a, v16h b, v8f c) {
  // D = A(16x32 f16) * B(32x16 f16) + C(16x16 f32)
  return __builtin_amdgcn_wmma_f32_16x16x32_f16(false, a, false, b, (short)0, c,
                                                false, false);
}

// Load one 16x32 f16 fragment from a row-major [rows, ld] matrix.
// ISA layout: lanes 0-15 -> M=lane, K pairs {kb..kb+7, kb+16..kb+23};
//             lanes 16-31 -> M=lane-16, K pairs {kb+8..kb+15, kb+24..kb+31}.
// Works for both A-tiles (rows = M) and B-tiles from W[N,K] (rows = N).
__device__ __forceinline__ v16h load_frag(const h16* base, int ld, int rowbase,
                                          int kb, int lane) {
  int l = lane & 15, hi = lane >> 4;
  const h16* p = base + (size_t)(rowbase + l) * (size_t)ld + kb + hi * 8;
  v8h lo  = *(const v8h*)(p);
  v8h hi8 = *(const v8h*)(p + 16);
  return __builtin_shufflevector(lo, hi8, 0, 1, 2, 3, 4, 5, 6, 7, 8, 9, 10, 11,
                                 12, 13, 14, 15);
}

// ---------------------------------------------------------------------------
// Elementwise / setup kernels
// ---------------------------------------------------------------------------
__global__ void k_zero(float* p, int n) {
  for (int i = blockIdx.x * blockDim.x + threadIdx.x; i < n;
       i += gridDim.x * blockDim.x)
    p[i] = 0.f;
}

__global__ void k_f2h(const float* __restrict__ in, h16* __restrict__ out, int n) {
  int i = blockIdx.x * blockDim.x + threadIdx.x;
  if (i < n) out[i] = (h16)in[i];
}

// torch Embedding(max_norm=1) renorm, one wave per row of 64
__global__ void k_renorm(const float* __restrict__ in, float* __restrict__ out) {
  int row = blockIdx.x, lane = threadIdx.x;
  float a = in[(size_t)row * 64 + lane];
  float b = in[(size_t)row * 64 + 32 + lane];
  float s = a * a + b * b;
  for (int m = 16; m >= 1; m >>= 1) s += __shfl_xor(s, m);
  float nrm = sqrtf(s);
  float sc = (nrm > 1.f) ? 1.f / fmaxf(nrm, 1e-7f) : 1.f;
  out[(size_t)row * 64 + lane] = a * sc;
  out[(size_t)row * 64 + 32 + lane] = b * sc;
}

// per-column BN (training, biased var) over [R,64] -> scale/shift
__global__ void k_colbn(const float* __restrict__ A, int R,
                        const float* __restrict__ gg, const float* __restrict__ bb,
                        float* __restrict__ sc, float* __restrict__ sh) {
  int h = threadIdx.x;
  float s = 0.f, q = 0.f;
  for (int r = 0; r < R; ++r) {
    float v = A[(size_t)r * 64 + h];
    s += v; q += v * v;
  }
  float mean = s / (float)R, var = q / (float)R - mean * mean;
  float scl = gg[h] * rsqrtf(var + 1e-5f);
  sc[h] = scl; sh[h] = bb[h] - mean * scl;
}

__global__ void k_bnfin(const float* __restrict__ S, const float* __restrict__ Q,
                        const float* __restrict__ gg, const float* __restrict__ bb,
                        float* __restrict__ sc, float* __restrict__ sh,
                        float invN, int C) {
  int c = blockIdx.x * blockDim.x + threadIdx.x;
  if (c >= C) return;
  float mean = S[c] * invN, var = Q[c] * invN - mean * mean;
  float scl = gg[c] * rsqrtf(var + 1e-5f);
  sc[c] = scl; sh[c] = bb[c] - mean * scl;
}

// gcn_norm degree accumulation (edge part; +1 self-loop applied at use)
__global__ void k_deg(const int* __restrict__ col, const float* __restrict__ w,
                      int E, float* __restrict__ deg) {
  int e = blockIdx.x * blockDim.x + threadIdx.x;
  if (e < E) atomicAdd(&deg[col[e]], w[e]);
}

// SGConv edge aggregation: agg[col] += dis[row]*w*dis[col] * x[row]
__global__ void k_sgedge(const int* __restrict__ ei, const float* __restrict__ w,
                         int E, const float* __restrict__ deg,
                         const float* __restrict__ xv, float* __restrict__ agg) {
  long long i = (long long)blockIdx.x * blockDim.x + threadIdx.x;
  if (i >= (long long)E * 64) return;
  int e = (int)(i >> 6), h = (int)(i & 63);
  int r = ei[e], c = ei[E + e];
  float nm = rsqrtf(deg[r] + 1.f) * w[e] * rsqrtf(deg[c] + 1.f);
  atomicAdd(&agg[(size_t)c * 64 + h], nm * xv[(size_t)r * 64 + h]);
}

// add self-loop term and convert to f16 GEMM input
__global__ void k_selfagg(const float* __restrict__ agg, const float* __restrict__ xv,
                          const float* __restrict__ deg, h16* __restrict__ out,
                          int rows) {
  int i = blockIdx.x * blockDim.x + threadIdx.x;
  if (i >= rows * 64) return;
  int r = i >> 6;
  out[i] = (h16)(agg[i] + xv[i] / (deg[r] + 1.f));
}

// etv2 input: base(g) + 0.2*pos(row), rows 0..G-1 = block0, G..2G-1 = generic
__global__ void k_build_etin(const float* __restrict__ geneN,
                             const float* __restrict__ sc, const float* __restrict__ sh,
                             const float* __restrict__ posout, h16* __restrict__ etin) {
  int i = blockIdx.x * blockDim.x + threadIdx.x;
  if (i >= 2 * GG * 64) return;
  int r = i >> 6, h = i & 63;
  int g = (r < GG) ? r : r - GG;
  float base = fmaxf(geneN[(size_t)g * 64 + h] * sc[h] + sh[h], 0.f);
  etin[i] = (h16)(base + 0.2f * posout[i]);
}

// bn+relu, f32 -> f16 (per-column affine, C columns)
__global__ void k_bnrelu_h(const float* __restrict__ in, const float* __restrict__ sc,
                           const float* __restrict__ sh, h16* __restrict__ out,
                           int n, int C) {
  int i = blockIdx.x * blockDim.x + threadIdx.x;
  if (i >= n) return;
  int h = i % C;
  out[i] = (h16)fmaxf(in[i] * sc[h] + sh[h], 0.f);
}

__global__ void k_pad_cgsw1(const float* __restrict__ w, h16* __restrict__ out) {
  int i = blockIdx.x * blockDim.x + threadIdx.x;
  if (i >= 64 * WPAD) return;
  int n = i / WPAD, k = i - n * WPAD;
  out[i] = (k < GG) ? (h16)w[(size_t)n * GG + k] : (h16)0.f;
}

__global__ void k_pad_ind2(const float* __restrict__ w2, h16* __restrict__ out) {
  int i = blockIdx.x * blockDim.x + threadIdx.x;
  if (i >= NPAD * 64) return;
  int g = i >> 6, k = i & 63;
  out[i] = (g < GG) ? (h16)w2[(size_t)g * 65 + 1 + k] : (h16)0.f;
}

// ---------------------------------------------------------------------------
// Generic wave-per-16-row-tile WMMA GEMM: out = A[M,K](f16) @ W[N,K]^T + bias
// Optional per-column weighted sum/sumsq accumulation (for batch-norm stats).
// M multiple of 16, K multiple of 32, N multiple of 16.
// ---------------------------------------------------------------------------
__global__ void __launch_bounds__(32) k_gemm(
    const h16* __restrict__ A, int lda, const h16* __restrict__ W, int ldw,
    const float* __restrict__ bias, int N, int K, float* __restrict__ outF,
    int ldo, int relu, float* __restrict__ statS, float* __restrict__ statQ,
    int wBound, float wlo, float whi) {
  int lane = threadIdx.x, l = lane & 15, hi = lane >> 4;
  int rowbase = blockIdx.x * 16;
  for (int nt = 0; nt < N; nt += 16) {
    v8f c = {0.f, 0.f, 0.f, 0.f, 0.f, 0.f, 0.f, 0.f};
    for (int kb = 0; kb < K; kb += 32) {
      v16h a = load_frag(A, lda, rowbase, kb, lane);
      v16h b = load_frag(W, ldw, nt, kb, lane);
      c = wmma16(a, b, c);
    }
    float bv = bias ? bias[nt + l] : 0.f;
    float s = 0.f, q = 0.f;
#pragma unroll
    for (int r = 0; r < 8; ++r) {
      int row = rowbase + r + 8 * hi;  // C layout: M = r + 8*(lane>>4), N = lane&15
      float v = c[r] + bv;
      if (relu) v = fmaxf(v, 0.f);
      if (outF) outF[(size_t)row * ldo + nt + l] = v;
      if (statS) {
        float wt = (row < wBound) ? wlo : whi;
        s += wt * v; q += wt * v * v;
      }
    }
    if (statS) {
      s += __shfl_xor(s, 16); q += __shfl_xor(q, 16);
      if (hi == 0) { atomicAdd(&statS[nt + l], s); atomicAdd(&statQ[nt + l], q); }
    }
  }
}

// ---------------------------------------------------------------------------
// pert-fuse MLP over B=64 rows (tiny, scalar): et = relu(bn-mlp(sum of pge))
// ---------------------------------------------------------------------------
__global__ void k_pf(const float* __restrict__ pge, const int* __restrict__ pidx,
                     const float* __restrict__ W1, const float* __restrict__ b1,
                     const float* __restrict__ gg, const float* __restrict__ bb,
                     const float* __restrict__ W2, const float* __restrict__ b2,
                     float* __restrict__ et) {
  __shared__ float sm[64 * 64];
  __shared__ float tm[64 * 64];
  int h = threadIdx.x;
  for (int r = 0; r < 64; ++r)
    sm[r * 64 + h] = pge[(size_t)pidx[2 * r] * 64 + h] +
                     pge[(size_t)pidx[2 * r + 1] * 64 + h];
  __syncthreads();
  for (int r = 0; r < 64; ++r) {
    float a = b1[h];
    for (int k = 0; k < 64; ++k) a += sm[r * 64 + k] * W1[h * 64 + k];
    tm[r * 64 + h] = a;
  }
  __syncthreads();
  float s = 0.f, q = 0.f;
  for (int r = 0; r < 64; ++r) { float v = tm[r * 64 + h]; s += v; q += v * v; }
  float mean = s / 64.f, var = q / 64.f - mean * mean;
  float sc = gg[h] * rsqrtf(var + 1e-5f), sh = bb[h] - mean * sc;
  for (int r = 0; r < 64; ++r) tm[r * 64 + h] = fmaxf(tm[r * 64 + h] * sc + sh, 0.f);
  __syncthreads();
  for (int r = 0; r < 64; ++r) {
    float a = b2[h];
    for (int k = 0; k < 64; ++k) a += tm[r * 64 + k] * W2[h * 64 + k];
    et[r * 64 + h] = fmaxf(a, 0.f);
  }
}

// ---------------------------------------------------------------------------
// bnp stats over virtual 320000-row batch, analytic decomposition:
// x[b,g,h] = etv[blk(b)][g][h] + et[b][h]  (block0 once, generic 63x)
// ---------------------------------------------------------------------------
__global__ void k_bnp(const float* __restrict__ etv, const float* __restrict__ et,
                      const float* __restrict__ gg, const float* __restrict__ bb,
                      float* __restrict__ sc, float* __restrict__ sh) {
  int h = threadIdx.x;
  float S0 = 0.f, SS0 = 0.f, Sg = 0.f, SSg = 0.f;
  for (int g = 0; g < GG; ++g) {
    float a = etv[(size_t)g * 64 + h];            S0 += a; SS0 += a * a;
    float c = etv[(size_t)(GG + g) * 64 + h];     Sg += c; SSg += c * c;
  }
  float se = 0.f, se2 = 0.f;
  for (int b = 0; b < BB; ++b) { float e = et[b * 64 + h]; se += e; se2 += e * e; }
  float e0 = et[h];
  const float N = (float)BB * (float)GG;
  float mean = (S0 + 63.f * Sg + (float)GG * se) / N;
  float ex2 = (SS0 + 63.f * SSg + 2.f * (e0 * S0 + (se - e0) * Sg) +
               (float)GG * se2) / N;
  float var = ex2 - mean * mean;
  float s = gg[h] * rsqrtf(var + 1e-5f);
  sc[h] = s; sh[h] = bb[h] - mean * s;
}

// ---------------------------------------------------------------------------
// rec MLP pass 1: reconstruct base2 tile on the fly, h1 = base2 @ W1^T + b1,
// accumulate per-column sum/sumsq (128 cols) for BN. One wave / 16 rows.
// ---------------------------------------------------------------------------
__global__ void __launch_bounds__(32) k_rec1(
    const float* __restrict__ etv, const float* __restrict__ et,
    const float* __restrict__ bsc, const float* __restrict__ bsh,
    const h16* __restrict__ W1h, const float* __restrict__ b1,
    float* __restrict__ statS, float* __restrict__ statQ) {
  __shared__ __align__(16) h16 At[16 * 64];
  int lane = threadIdx.x;
  int rowbase = blockIdx.x * 16;
  for (int i = lane; i < 1024; i += 32) {
    int r = i >> 6, h = i & 63;
    int grow = rowbase + r, b = grow / GG, g = grow - b * GG;
    float v = (etv[(size_t)((b == 0) ? g : GG + g) * 64 + h] + et[b * 64 + h]) *
                  bsc[h] + bsh[h];
    At[i] = (h16)fmaxf(v, 0.f);
  }
  __syncthreads();
  int l = lane & 15, hi = lane >> 4;
  for (int nt = 0; nt < 128; nt += 16) {
    v8f c = {0.f, 0.f, 0.f, 0.f, 0.f, 0.f, 0.f, 0.f};
#pragma unroll
    for (int kb = 0; kb < 64; kb += 32) {
      v16h a = load_frag(At, 64, 0, kb, lane);
      v16h b = load_frag(W1h, 64, nt, kb, lane);
      c = wmma16(a, b, c);
    }
    float bv = b1[nt + l], s = 0.f, q = 0.f;
#pragma unroll
    for (int r = 0; r < 8; ++r) { float v = c[r] + bv; s += v; q += v * v; }
    s += __shfl_xor(s, 16); q += __shfl_xor(q, 16);
    if (hi == 0) { atomicAdd(&statS[nt + l], s); atomicAdd(&statQ[nt + l], q); }
  }
}

// ---------------------------------------------------------------------------
// rec MLP pass 2 + per-gene decoder 1, fully fused in LDS.
// ---------------------------------------------------------------------------
__global__ void __launch_bounds__(32) k_rec2(
    const float* __restrict__ etv, const float* __restrict__ et,
    const float* __restrict__ bsc, const float* __restrict__ bsh,
    const h16* __restrict__ W1h, const float* __restrict__ b1,
    const float* __restrict__ rsc, const float* __restrict__ rsh,
    const h16* __restrict__ W2h, const float* __restrict__ b2,
    const float* __restrict__ iw1, const float* __restrict__ ib1,
    float* __restrict__ wout, h16* __restrict__ whout) {
  __shared__ __align__(16) h16 At[16 * 64];
  __shared__ __align__(16) h16 H1[16 * 128];
  __shared__ float RC[16 * 64];
  int lane = threadIdx.x;
  int rowbase = blockIdx.x * 16;
  for (int i = lane; i < 1024; i += 32) {
    int r = i >> 6, h = i & 63;
    int grow = rowbase + r, b = grow / GG, g = grow - b * GG;
    float v = (etv[(size_t)((b == 0) ? g : GG + g) * 64 + h] + et[b * 64 + h]) *
                  bsc[h] + bsh[h];
    At[i] = (h16)fmaxf(v, 0.f);
  }
  __syncthreads();
  int l = lane & 15, hi = lane >> 4;
  for (int nt = 0; nt < 128; nt += 16) {
    v8f c = {0.f, 0.f, 0.f, 0.f, 0.f, 0.f, 0.f, 0.f};
#pragma unroll
    for (int kb = 0; kb < 64; kb += 32) {
      v16h a = load_frag(At, 64, 0, kb, lane);
      v16h b = load_frag(W1h, 64, nt, kb, lane);
      c = wmma16(a, b, c);
    }
    int col = nt + l;
    float scl = rsc[col], shf = rsh[col], bv = b1[col];
#pragma unroll
    for (int r = 0; r < 8; ++r) {
      float v = fmaxf((c[r] + bv) * scl + shf, 0.f);
      H1[(r + 8 * hi) * 128 + col] = (h16)v;
    }
  }
  __syncthreads();
  for (int nt = 0; nt < 64; nt += 16) {
    v8f c = {0.f, 0.f, 0.f, 0.f, 0.f, 0.f, 0.f, 0.f};
#pragma unroll
    for (int kb = 0; kb < 128; kb += 32) {
      v16h a = load_frag(H1, 128, 0, kb, lane);
      v16h b = load_frag(W2h, 128, nt, kb, lane);
      c = wmma16(a, b, c);
    }
    int col = nt + l;
    float bv = b2[col];
#pragma unroll
    for (int r = 0; r < 8; ++r) RC[(r + 8 * hi) * 64 + col] = c[r] + bv;  // no relu
  }
  __syncthreads();
  {  // w[b,g] = dot(rec_row, indv_w1[g]) + indv_b1[g]; lanes split K in halves
    int row = l, half = hi;
    int grow = rowbase + row, b = grow / GG, g = grow - b * GG;
    float acc = 0.f;
    for (int k = half * 32; k < half * 32 + 32; ++k)
      acc += RC[row * 64 + k] * iw1[(size_t)g * 64 + k];
    acc += __shfl_xor(acc, 16);
    if (half == 0) {
      float wv = acc + ib1[g];
      wout[(size_t)b * GG + g] = wv;
      whout[(size_t)b * WPAD + g] = (h16)wv;
    }
  }
}

// cross-gene tail: bn+relu over 64 rows then @ cgs_w2^T + b2 (last_relu=False)
__global__ void k_cgs_tail(const float* __restrict__ t1, const float* __restrict__ S,
                           const float* __restrict__ Q, const float* __restrict__ gg,
                           const float* __restrict__ bb, const float* __restrict__ W2,
                           const float* __restrict__ b2, h16* __restrict__ cgeh) {
  __shared__ float hm[64 * 64];
  int h = threadIdx.x;
  float mean = S[h] / 64.f, var = Q[h] / 64.f - mean * mean;
  float sc = gg[h] * rsqrtf(var + 1e-5f), sh = bb[h] - mean * sc;
  for (int r = 0; r < 64; ++r)
    hm[r * 64 + h] = fmaxf(t1[r * 64 + h] * sc + sh, 0.f);
  __syncthreads();
  for (int r = 0; r < 64; ++r) {
    float a = b2[h];
    for (int k = 0; k < 64; ++k) a += hm[r * 64 + k] * W2[h * 64 + k];
    cgeh[r * 64 + h] = (h16)a;
  }
}

// final: out[b,g] = (cge @ ind2[:,1:]^T)[b,g] + w*ind2[:,0] + indv_b2 + x
__global__ void __launch_bounds__(32) k_final(
    const h16* __restrict__ cgeh, const h16* __restrict__ ind2h,
    const float* __restrict__ wbuf, const float* __restrict__ iw2,
    const float* __restrict__ ib2, const float* __restrict__ xin,
    float* __restrict__ outp) {
  int lane = threadIdx.x, l = lane & 15, hi = lane >> 4;
  int ntile = blockIdx.x * 16, rtile = blockIdx.y * 16;
  v8f c = {0.f, 0.f, 0.f, 0.f, 0.f, 0.f, 0.f, 0.f};
#pragma unroll
  for (int kb = 0; kb < 64; kb += 32) {
    v16h a = load_frag(cgeh, 64, rtile, kb, lane);
    v16h b = load_frag(ind2h, 64, ntile, kb, lane);
    c = wmma16(a, b, c);
  }
  int g = ntile + l;
  if (g < GG) {
    float c0 = iw2[(size_t)g * 65], bv = ib2[g];
#pragma unroll
    for (int r = 0; r < 8; ++r) {
      int b = rtile + r + 8 * hi;
      size_t idx = (size_t)b * GG + g;
      outp[idx] = c[r] + wbuf[idx] * c0 + bv + xin[idx];
    }
  }
}

// ---------------------------------------------------------------------------
extern "C" void kernel_launch(void* const* d_in, const int* in_sizes, int n_in,
                              void* d_out, int out_size, void* d_ws, size_t ws_size,
                              hipStream_t stream) {
  (void)in_sizes; (void)n_in; (void)out_size; (void)ws_size;
  const int Ec = 20 * GG, Es = 20 * PP, TG = 2 * GG;

  const float* xin       = (const float*)d_in[0];
  const int*   pert_idx  = (const int*)d_in[1];
  const int*   coex_ei   = (const int*)d_in[2];
  const float* coex_w    = (const float*)d_in[3];
  const int*   sim_ei    = (const int*)d_in[4];
  const float* sim_w     = (const float*)d_in[5];
  const float* gene_emb_w= (const float*)d_in[6];
  const float* emb_pos_w = (const float*)d_in[7];
  const float* pert_emb_w= (const float*)d_in[8];
  const float* bn_emb_g  = (const float*)d_in[9];
  const float* bn_emb_b  = (const float*)d_in[10];
  const float* pos_conv_w= (const float*)d_in[11];
  const float* pos_conv_b= (const float*)d_in[12];
  const float* sim_conv_w= (const float*)d_in[13];
  const float* sim_conv_b= (const float*)d_in[14];
  const float* etv2_w1   = (const float*)d_in[15];
  const float* etv2_b1   = (const float*)d_in[16];
  const float* etv2_bng  = (const float*)d_in[17];
  const float* etv2_bnb  = (const float*)d_in[18];
  const float* etv2_w2   = (const float*)d_in[19];
  const float* etv2_b2   = (const float*)d_in[20];
  const float* pf_w1     = (const float*)d_in[21];
  const float* pf_b1     = (const float*)d_in[22];
  const float* pf_bng    = (const float*)d_in[23];
  const float* pf_bnb    = (const float*)d_in[24];
  const float* pf_w2     = (const float*)d_in[25];
  const float* pf_b2     = (const float*)d_in[26];
  const float* bnp_g     = (const float*)d_in[27];
  const float* bnp_b     = (const float*)d_in[28];
  const float* rec_w1    = (const float*)d_in[29];
  const float* rec_b1    = (const float*)d_in[30];
  const float* rec_bng   = (const float*)d_in[31];
  const float* rec_bnb   = (const float*)d_in[32];
  const float* rec_w2    = (const float*)d_in[33];
  const float* rec_b2    = (const float*)d_in[34];
  const float* indv_w1   = (const float*)d_in[35];
  const float* indv_b1   = (const float*)d_in[36];
  const float* cgs_w1    = (const float*)d_in[37];
  const float* cgs_b1    = (const float*)d_in[38];
  const float* cgs_bng   = (const float*)d_in[39];
  const float* cgs_bnb   = (const float*)d_in[40];
  const float* cgs_w2    = (const float*)d_in[41];
  const float* cgs_b2    = (const float*)d_in[42];
  const float* indv_w2   = (const float*)d_in[43];
  const float* indv_b2   = (const float*)d_in[44];
  float* outp = (float*)d_out;

  char* ws = (char*)d_ws; size_t off = 0;
  auto alloc = [&](size_t bytes) -> void* {
    off = (off + 255) & ~(size_t)255;
    void* p = ws + off; off += bytes; return p;
  };

  // ---- zero zone (atomic accumulators + K-padded w matrix) ----
  float* deg_coex = (float*)alloc((size_t)GG * 4);
  float* deg_sim  = (float*)alloc((size_t)PP * 4);
  float* agg0     = (float*)alloc((size_t)GG * 64 * 4);
  float* agg_sim  = (float*)alloc((size_t)PP * 64 * 4);
  float* etv_stat = (float*)alloc(128 * 4);
  float* rec_stat = (float*)alloc(256 * 4);
  float* cgs_stat = (float*)alloc(128 * 4);
  h16*   wh       = (h16*)alloc((size_t)BB * WPAD * 2);
  size_t zbytes = (off + 3) & ~(size_t)3;
  int zeroFloats = (int)(zbytes / 4);

  // ---- other scratch ----
  float* geneN  = (float*)alloc((size_t)GG * 64 * 4);
  float* posN   = (float*)alloc((size_t)GG * 64 * 4);
  float* pertN  = (float*)alloc((size_t)PP * 64 * 4);
  h16*   aggh   = (h16*)alloc((size_t)TG * 64 * 2);
  h16*   aggsh  = (h16*)alloc((size_t)PP * 64 * 2);
  float* posout = (float*)alloc((size_t)TG * 64 * 4);
  h16*   etin   = (h16*)alloc((size_t)TG * 64 * 2);
  float* h1     = (float*)alloc((size_t)TG * 64 * 4);
  h16*   h1h    = (h16*)alloc((size_t)TG * 64 * 2);
  float* etv    = (float*)alloc((size_t)TG * 64 * 4);
  float* pge    = (float*)alloc((size_t)PP * 64 * 4);
  float* et     = (float*)alloc(64 * 64 * 4);
  float* sc_emb = (float*)alloc(64 * 4);
  float* sh_emb = (float*)alloc(64 * 4);
  float* etv_sc = (float*)alloc(64 * 4);
  float* etv_sh = (float*)alloc(64 * 4);
  float* rec_sc = (float*)alloc(128 * 4);
  float* rec_sh = (float*)alloc(128 * 4);
  float* bnp_sc = (float*)alloc(64 * 4);
  float* bnp_sh = (float*)alloc(64 * 4);
  h16* posWh  = (h16*)alloc(4096 * 2);
  h16* simWh  = (h16*)alloc(4096 * 2);
  h16* e1Wh   = (h16*)alloc(4096 * 2);
  h16* e2Wh   = (h16*)alloc(4096 * 2);
  h16* recW1h = (h16*)alloc(8192 * 2);
  h16* recW2h = (h16*)alloc(8192 * 2);
  h16* cgsW1h = (h16*)alloc((size_t)64 * WPAD * 2);
  h16* ind2h  = (h16*)alloc((size_t)NPAD * 64 * 2);
  float* t1   = (float*)alloc(64 * 64 * 4);
  h16* cgeh   = (h16*)alloc(4096 * 2);
  float* wbuf = (float*)alloc((size_t)BB * GG * 4);

  // ---- pipeline ----
  k_zero<<<1024, 256, 0, stream>>>((float*)ws, zeroFloats);

  k_renorm<<<GG, 32, 0, stream>>>(gene_emb_w, geneN);
  k_renorm<<<GG, 32, 0, stream>>>(emb_pos_w, posN);
  k_renorm<<<PP, 32, 0, stream>>>(pert_emb_w, pertN);
  k_colbn<<<1, 64, 0, stream>>>(geneN, GG, bn_emb_g, bn_emb_b, sc_emb, sh_emb);

  k_deg<<<(Ec + 255) / 256, 256, 0, stream>>>(coex_ei + Ec, coex_w, Ec, deg_coex);
  k_deg<<<(Es + 255) / 256, 256, 0, stream>>>(sim_ei + Es, sim_w, Es, deg_sim);
  k_sgedge<<<(Ec * 64 + 255) / 256, 256, 0, stream>>>(coex_ei, coex_w, Ec, deg_coex, posN, agg0);
  k_sgedge<<<(Es * 64 + 255) / 256, 256, 0, stream>>>(sim_ei, sim_w, Es, deg_sim, pertN, agg_sim);

  k_f2h<<<16, 256, 0, stream>>>(pos_conv_w, posWh, 4096);
  k_f2h<<<16, 256, 0, stream>>>(sim_conv_w, simWh, 4096);
  k_f2h<<<16, 256, 0, stream>>>(etv2_w1, e1Wh, 4096);
  k_f2h<<<16, 256, 0, stream>>>(etv2_w2, e2Wh, 4096);
  k_f2h<<<32, 256, 0, stream>>>(rec_w1, recW1h, 8192);
  k_f2h<<<32, 256, 0, stream>>>(rec_w2, recW2h, 8192);
  k_pad_cgsw1<<<(64 * WPAD + 255) / 256, 256, 0, stream>>>(cgs_w1, cgsW1h);
  k_pad_ind2<<<(NPAD * 64 + 255) / 256, 256, 0, stream>>>(indv_w2, ind2h);

  // SGConv inputs: block-0 rows get edge agg + self-loop; generic rows self only
  k_selfagg<<<(GG * 64 + 255) / 256, 256, 0, stream>>>(agg0, posN, deg_coex, aggh, GG);
  k_f2h<<<(GG * 64 + 255) / 256, 256, 0, stream>>>(posN, aggh + (size_t)GG * 64, GG * 64);
  k_selfagg<<<(PP * 64 + 255) / 256, 256, 0, stream>>>(agg_sim, pertN, deg_sim, aggsh, PP);

  // pos conv + sim conv (WMMA)
  k_gemm<<<TG / 16, 32, 0, stream>>>(aggh, 64, posWh, 64, pos_conv_b, 64, 64,
                                     posout, 64, 0, nullptr, nullptr, 0, 1.f, 1.f);
  k_gemm<<<PP / 16, 32, 0, stream>>>(aggsh, 64, simWh, 64, sim_conv_b, 64, 64,
                                     pge, 64, 0, nullptr, nullptr, 0, 1.f, 1.f);

  k_build_etin<<<(TG * 64 + 255) / 256, 256, 0, stream>>>(geneN, sc_emb, sh_emb, posout, etin);

  // etv2 MLP (weighted BN stats: block0 rows x1, generic rows x63)
  k_gemm<<<TG / 16, 32, 0, stream>>>(etin, 64, e1Wh, 64, etv2_b1, 64, 64, h1, 64,
                                     0, etv_stat, etv_stat + 64, GG, 1.f, 63.f);
  k_bnfin<<<1, 64, 0, stream>>>(etv_stat, etv_stat + 64, etv2_bng, etv2_bnb,
                                etv_sc, etv_sh, 1.f / 320000.f, 64);
  k_bnrelu_h<<<(TG * 64 + 255) / 256, 256, 0, stream>>>(h1, etv_sc, etv_sh, h1h, TG * 64, 64);
  k_gemm<<<TG / 16, 32, 0, stream>>>(h1h, 64, e2Wh, 64, etv2_b2, 64, 64, etv, 64,
                                     1, nullptr, nullptr, 0, 1.f, 1.f);

  // perturbation path
  k_pf<<<1, 64, 0, stream>>>(pge, pert_idx, pf_w1, pf_b1, pf_bng, pf_bnb,
                             pf_w2, pf_b2, et);
  k_bnp<<<1, 64, 0, stream>>>(etv, et, bnp_g, bnp_b, bnp_sc, bnp_sh);

  // rec MLP over 320000 virtual rows (fused base2 reconstruction)
  k_rec1<<<(BB * GG) / 16, 32, 0, stream>>>(etv, et, bnp_sc, bnp_sh, recW1h,
                                            rec_b1, rec_stat, rec_stat + 128);
  k_bnfin<<<2, 64, 0, stream>>>(rec_stat, rec_stat + 128, rec_bng, rec_bnb,
                                rec_sc, rec_sh, 1.f / 320000.f, 128);
  k_rec2<<<(BB * GG) / 16, 32, 0, stream>>>(etv, et, bnp_sc, bnp_sh, recW1h,
                                            rec_b1, rec_sc, rec_sh, recW2h, rec_b2,
                                            indv_w1, indv_b1, wbuf, wh);

  // cross-gene state: t1 = w @ cgs_w1^T (K=5024, WMMA) then tiny tail
  k_gemm<<<BB / 16, 32, 0, stream>>>(wh, WPAD, cgsW1h, WPAD, cgs_b1, 64, WPAD,
                                     t1, 64, 0, cgs_stat, cgs_stat + 64,
                                     1 << 30, 1.f, 1.f);
  k_cgs_tail<<<1, 64, 0, stream>>>(t1, cgs_stat, cgs_stat + 64, cgs_bng, cgs_bnb,
                                   cgs_w2, cgs_b2, cgeh);

  // final decoder with fused residual epilogue
  k_final<<<dim3(NPAD / 16, BB / 16), 32, 0, stream>>>(cgeh, ind2h, wbuf, indv_w2,
                                                       indv_b2, xin, outp);
}
